// Qwen3Attention_7000796692872
// MI455X (gfx1250) — matrix operand order repaired and checked by reading, not compiled, not run on hardware
//
#include <hip/hip_runtime.h>
#include <hip/hip_bf16.h>

#define S_LEN 2048
#define HID   4096
#define NQ    32
#define NKV   8
#define HD    128
#define QKVD  6144   // (NQ + 2*NKV) * HD

#define KSTEP  64
#define LDSROW 72    // 64 data + 8 pad (bf16 elems) per LDS tile row; 144B, 16B-aligned
#define PLANE_BYTES (128 * LDSROW * 2)      // 18432 B per tile plane
#define BUF_BYTES   (2 * PLANE_BYTES)       // A+B planes per buffer

typedef __bf16 bf16;
typedef __attribute__((ext_vector_type(16))) __bf16 bf16x16;
typedef __attribute__((ext_vector_type(8)))  __bf16 bf16x8;
typedef __attribute__((ext_vector_type(8)))  float  f32x8;
typedef __attribute__((ext_vector_type(4)))  unsigned int u32x4;
typedef __attribute__((ext_vector_type(8)))  int i32x8;
typedef __attribute__((ext_vector_type(4)))  int i32x4;

#define HAVE_TDM 0
#if defined(__has_builtin)
#if __has_builtin(__builtin_amdgcn_tensor_load_to_lds)
#undef HAVE_TDM
#define HAVE_TDM 1
#endif
#endif

union ABFrag { bf16x16 v; bf16x8 h[2]; };

__device__ inline f32x8 wmma_bf16(const bf16x16& a, const bf16x16& b, const f32x8& c) {
  // (neg_a, A, neg_b, B, c_mod, C, reuse_a, reuse_b)
  return __builtin_amdgcn_wmma_f32_16x16x32_bf16(false, a, false, b, (short)0, c, false, false);
}

#if HAVE_TDM
// TDM 2D tile load: tile_d0 x tile_d1 elements (2B each) from a row-major tensor
// (row stride = stride0 elements) into LDS at byte offset ldsOff, padding the LDS
// destination by 4 DWORDs every 32 DWORDs (=> LDS row stride 72 bf16).
// D# bitfields per CDNA5 ISA.
__device__ inline void tdm_load_2d(unsigned ldsOff, const bf16* gptr,
                                   int tensor_d0, int tensor_d1, long long stride0,
                                   int tile_d0, int tile_d1) {
  unsigned long long ga = (unsigned long long)(uintptr_t)gptr;
  u32x4 g0;
  g0[0] = 1u;                                       // count=1, user descriptor
  g0[1] = ldsOff;                                   // lds_addr (bytes)
  g0[2] = (unsigned)(ga & 0xffffffffull);           // global_addr[31:0]
  g0[3] = (unsigned)((ga >> 32) & 0x01ffffffull)    // global_addr[56:32]
        | 0x80000000u;                              // type=2 ("image") in [127:126]
  i32x8 g1;
  g1[0] = (int)(0x00010000u      // data_size = 1 -> 2 bytes
              | (1u << 20)       // pad_enable
              | (4u << 22)       // pad_interval code 4 -> every 32 DWORDs
              | (3u << 25));     // pad_amount  code 3 -> 4 DWORDs (8 bf16)
  g1[1] = (int)(((unsigned)tensor_d0 & 0xffffu) << 16);              // tensor_dim0[15:0]
  g1[2] = (int)((((unsigned)tensor_d0 >> 16) & 0xffffu)
              | (((unsigned)tensor_d1 & 0xffffu) << 16));            // dim0 hi | dim1 lo
  g1[3] = (int)((((unsigned)tensor_d1 >> 16) & 0xffffu)
              | (((unsigned)tile_d0 & 0xffffu) << 16));              // dim1 hi | tile_dim0
  g1[4] = (int)((unsigned)tile_d1 & 0xffffu);                        // tile_dim1 (tile_dim2=0)
  g1[5] = (int)(unsigned)(stride0 & 0xffffffffll);                   // tensor_dim0_stride lo
  g1[6] = (int)(unsigned)((stride0 >> 32) & 0xffffll);               // stride hi (stride1=0)
  g1[7] = 0;
  i32x4 z4 = {0, 0, 0, 0};
#if __clang_major__ >= 23
  i32x8 z8 = {0, 0, 0, 0, 0, 0, 0, 0};
  __builtin_amdgcn_tensor_load_to_lds(g0, g1, z4, z4, z8, 0);
#else
  __builtin_amdgcn_tensor_load_to_lds(g0, g1, z4, z4, 0);
#endif
}
#endif

// ---------------------------------------------------------------------------
// fp32 -> bf16 elementwise convert
// ---------------------------------------------------------------------------
__global__ void cvt_bf16(const float* __restrict__ in, bf16* __restrict__ out, size_t n) {
  size_t i = (size_t)blockIdx.x * blockDim.x + threadIdx.x;
  if (i < n) out[i] = (bf16)in[i];
}

// ---------------------------------------------------------------------------
// fp32 (K x N) -> bf16 transposed (N x K), tiled through LDS
// ---------------------------------------------------------------------------
__global__ __launch_bounds__(256)
void transpose_cvt(const float* __restrict__ in, bf16* __restrict__ out, int K, int N) {
  __shared__ float tile[32][33];
  int nb = blockIdx.x * 32, kb = blockIdx.y * 32;
  int tx = threadIdx.x, ty = threadIdx.y;   // 32 x 8
  for (int i = 0; i < 32; i += 8)
    tile[ty + i][tx] = in[(size_t)(kb + ty + i) * N + nb + tx];
  __syncthreads();
  for (int i = 0; i < 32; i += 8)
    out[(size_t)(nb + ty + i) * K + kb + tx] = (bf16)tile[tx][ty + i];
}

// ---------------------------------------------------------------------------
// C[M][N] (fp32) = A[M][K] (bf16 row-major) * Bt[N][K]^T (bf16 row-major)
// 256 threads = 8 waves (2x4). Block tile 128x128, K-step 64, wave tile 64x32.
// Tiles DMA'd by the Tensor Data Mover (double-buffered, TENSORcnt-tracked);
// cooperative vector loads as fallback when the TDM builtin is unavailable.
// ---------------------------------------------------------------------------
__global__ __launch_bounds__(256)
void gemm_bf16(const bf16* __restrict__ A, const bf16* __restrict__ Bt,
               float* __restrict__ C, int M, int N, int K) {
  __shared__ __align__(16) bf16 smem[2 * 2 * 128 * LDSROW];   // [buf][plane][128][72]

  const int tid  = threadIdx.x;
  const int wid  = tid >> 5;
  const int lane = tid & 31;
  const int l16  = lane & 15;
  const int lh   = lane >> 4;
  const int mblk = blockIdx.y * 128;
  const int nblk = blockIdx.x * 128;
  const int wm   = (wid >> 2) * 64;        // 0 / 64
  const int wn   = (wid & 3) * 32;         // 0..96

  f32x8 acc[4][2] = {};

#if HAVE_TDM
  const unsigned ldsBase = (unsigned)(uintptr_t)smem;
  if (wid == 0)
    tdm_load_2d(ldsBase, A + (size_t)mblk * K, K, M, K, KSTEP, 128);
  else if (wid == 1)
    tdm_load_2d(ldsBase + PLANE_BYTES, Bt + (size_t)nblk * K, K, N, K, KSTEP, 128);
#endif

  int p = 0;
  for (int k0 = 0; k0 < K; k0 += KSTEP, p ^= 1) {
    bf16* Abuf = smem + p * (BUF_BYTES / 2);       // elements: p * 2*128*72
    bf16* Bbuf = Abuf + 128 * LDSROW;

#if HAVE_TDM
    if (wid == 0) {
      if (k0 + KSTEP < K) {
        tdm_load_2d(ldsBase + (p ^ 1) * BUF_BYTES,
                    A + (size_t)mblk * K + (k0 + KSTEP), K, M, K, KSTEP, 128);
        __builtin_amdgcn_s_wait_tensorcnt(1);      // in-order: tile k0 complete
      } else {
        __builtin_amdgcn_s_wait_tensorcnt(0);
      }
    } else if (wid == 1) {
      if (k0 + KSTEP < K) {
        tdm_load_2d(ldsBase + PLANE_BYTES + (p ^ 1) * BUF_BYTES,
                    Bt + (size_t)nblk * K + (k0 + KSTEP), K, N, K, KSTEP, 128);
        __builtin_amdgcn_s_wait_tensorcnt(1);
      } else {
        __builtin_amdgcn_s_wait_tensorcnt(0);
      }
    }
#else
    {
      int r = tid >> 1;                    // 0..127
      int c = (tid & 1) * 32;              // 0 / 32
      for (int u = 0; u < 4; u++)
        *(bf16x8*)&Abuf[r * LDSROW + c + u * 8] =
            *(const bf16x8*)&A[(size_t)(mblk + r) * K + k0 + c + u * 8];
      for (int u = 0; u < 4; u++)
        *(bf16x8*)&Bbuf[r * LDSROW + c + u * 8] =
            *(const bf16x8*)&Bt[(size_t)(nblk + r) * K + k0 + c + u * 8];
    }
#endif
    __syncthreads();

    for (int kk = 0; kk < 2; kk++) {
      ABFrag a[4], b[2];
      for (int i = 0; i < 4; i++) {
        const bf16* ar = Abuf + (wm + i * 16 + l16) * LDSROW + kk * 32;
        a[i].h[0] = *(const bf16x8*)&ar[lh * 8];
        a[i].h[1] = *(const bf16x8*)&ar[16 + lh * 8];
      }
      for (int j = 0; j < 2; j++) {
        const bf16* br = Bbuf + (wn + j * 16 + l16) * LDSROW + kk * 32;
        b[j].h[0] = *(const bf16x8*)&br[lh * 8];
        b[j].h[1] = *(const bf16x8*)&br[16 + lh * 8];
      }
      for (int i = 0; i < 4; i++)
        for (int j = 0; j < 2; j++)
          acc[i][j] = wmma_bf16(a[i].v, b[j].v, acc[i][j]);
    }
    __syncthreads();
  }

  for (int i = 0; i < 4; i++)
    for (int j = 0; j < 2; j++)
      for (int r = 0; r < 8; r++) {
        int row = mblk + wm + i * 16 + (lh ? 8 + r : r);
        int col = nblk + wn + j * 16 + l16;
        C[(size_t)row * N + col] = acc[i][j][r];
      }
}

// ---------------------------------------------------------------------------
// RMSNorm + RoPE for q/k heads; V copied transposed per KV head (Vt[g][d][s]).
// grid = (S, NQ + 2*NKV), 128 threads (one per head-dim element).
// ---------------------------------------------------------------------------
__global__ __launch_bounds__(128)
void rms_rope(const float* __restrict__ qkv,
              const float* __restrict__ qg, const float* __restrict__ kg,
              const float* __restrict__ cosb, const float* __restrict__ sinb,
              bf16* __restrict__ Q, bf16* __restrict__ Kc, bf16* __restrict__ Vt) {
  const int s  = blockIdx.x;
  const int hy = blockIdx.y;
  const int d  = threadIdx.x;
  __shared__ float red[4];

  if (hy >= NQ + NKV) {                    // V heads: transpose copy
    int g = hy - (NQ + NKV);
    float v = qkv[(size_t)s * QKVD + (size_t)(NQ + NKV) * HD + g * HD + d];
    Vt[((size_t)(g * HD + d)) * S_LEN + s] = (bf16)v;
    return;
  }
  const bool isq = hy < NQ;
  const int off  = isq ? hy * HD : NQ * HD + (hy - NQ) * HD;
  const float* gamma = isq ? qg : kg;

  float x = qkv[(size_t)s * QKVD + off + d];
  float ss = x * x;
  for (int m = 16; m >= 1; m >>= 1) ss += __shfl_xor(ss, m, 32);
  if ((threadIdx.x & 31) == 0) red[threadIdx.x >> 5] = ss;
  __syncthreads();
  float rms = rsqrtf((red[0] + red[1] + red[2] + red[3]) * (1.0f / HD) + 1e-6f);

  int dp   = (d < 64) ? d + 64 : d - 64;
  float xp = qkv[(size_t)s * QKVD + off + dp];
  float xn = x  * rms * gamma[d];
  float pn = xp * rms * gamma[dp];
  int cd   = (d < 64) ? d : d - 64;
  float c  = cosb[s * 64 + cd];
  float sn = sinb[s * 64 + cd];
  float o  = (d < 64) ? (xn * c - pn * sn) : (xn * c + pn * sn);

  if (isq) Q [(size_t)s * (NQ  * HD) + hy * HD + d]        = (bf16)o;
  else     Kc[(size_t)s * (NKV * HD) + (hy - NQ) * HD + d] = (bf16)o;
}

// ---------------------------------------------------------------------------
// Causal GQA flash attention. grid = (S/64, NQ), 128 threads = 4 waves,
// each wave owns 16 query rows. Key blocks of 32 (one WMMA K step).
// ---------------------------------------------------------------------------
__global__ __launch_bounds__(128)
void flash_attn(const bf16* __restrict__ Q, const bf16* __restrict__ Kc,
                const bf16* __restrict__ Vt, bf16* __restrict__ O) {
  __shared__ bf16 pbuf[4][16][32];         // per-wave P staging (C-layout -> A-frag)
  const int tid = threadIdx.x, wid = tid >> 5, lane = tid & 31;
  const int l16 = lane & 15, lh = lane >> 4;
  const int h = blockIdx.y, g = h >> 2;
  const int qbase = blockIdx.x * 64 + wid * 16;
  const float scale = 0.08838834764831845f;   // 1/sqrt(128)

  ABFrag qa[4];                            // Q row held in registers: 4 K-chunks of 32
  {
    const bf16* qrow = Q + (size_t)(qbase + l16) * (NQ * HD) + h * HD;
    for (int c = 0; c < 4; c++) {
      qa[c].h[0] = *(const bf16x8*)&qrow[c * 32 + lh * 8];
      qa[c].h[1] = *(const bf16x8*)&qrow[c * 32 + 16 + lh * 8];
    }
  }

  f32x8 accO[8] = {};
  float mrow[8], lrow[8];
  for (int i = 0; i < 8; i++) { mrow[i] = -3.0e38f; lrow[i] = 0.0f; }

  const int nkb = ((qbase + 15) >> 5) + 1;     // only lower triangle
  for (int kb = 0; kb < nkb; kb++) {
    const int tbase = kb * 32;

    f32x8 sc[2] = {};
    for (int j = 0; j < 2; j++) {
      const bf16* krow = Kc + (size_t)(tbase + j * 16 + l16) * (NKV * HD) + g * HD;
      for (int c = 0; c < 4; c++) {
        ABFrag kf;
        kf.h[0] = *(const bf16x8*)&krow[c * 32 + lh * 8];
        kf.h[1] = *(const bf16x8*)&krow[c * 32 + 16 + lh * 8];
        sc[j] = wmma_bf16(qa[c].v, kf.v, sc[j]);
      }
    }

    // scale + causal mask + online softmax (rows split across lane halves)
    float pm[2][8];
    for (int i = 0; i < 8; i++) {
      int srow = qbase + (lh ? 8 + i : i);
      float mx = -3.0e38f;
      for (int j = 0; j < 2; j++) {
        int t = tbase + j * 16 + l16;
        float v = sc[j][i] * scale;
        if (t > srow) v = -3.0e38f;
        pm[j][i] = v;
        mx = fmaxf(mx, v);
      }
      for (int m = 1; m < 16; m <<= 1) mx = fmaxf(mx, __shfl_xor(mx, m, 32));
      float newm = fmaxf(mrow[i], mx);
      float sfac = __expf(mrow[i] - newm);
      float rsum = 0.0f;
      for (int j = 0; j < 2; j++) {
        float pv = __expf(pm[j][i] - newm);
        pm[j][i] = pv;
        rsum += pv;
      }
      for (int m = 1; m < 16; m <<= 1) rsum += __shfl_xor(rsum, m, 32);
      lrow[i] = lrow[i] * sfac + rsum;
      mrow[i] = newm;
      for (int n = 0; n < 8; n++) accO[n][i] *= sfac;
    }

    // reshape P: C-layout -> LDS -> A-fragment (same-wave LDS is in-order)
    for (int i = 0; i < 8; i++) {
      int row = lh ? 8 + i : i;
      pbuf[wid][row][l16]      = (bf16)pm[0][i];
      pbuf[wid][row][16 + l16] = (bf16)pm[1][i];
    }
    ABFrag pa;
    pa.h[0] = *(const bf16x8*)&pbuf[wid][l16][lh * 8];
    pa.h[1] = *(const bf16x8*)&pbuf[wid][l16][16 + lh * 8];

    for (int n = 0; n < 8; n++) {          // P(16x32) x V(32x16) per d-tile
      ABFrag vf;
      const bf16* vr = Vt + (size_t)(g * HD + n * 16 + l16) * S_LEN + tbase;
      vf.h[0] = *(const bf16x8*)&vr[lh * 8];
      vf.h[1] = *(const bf16x8*)&vr[16 + lh * 8];
      accO[n] = wmma_bf16(pa.v, vf.v, accO[n]);
    }
  }

  for (int n = 0; n < 8; n++)
    for (int i = 0; i < 8; i++) {
      int row = qbase + (lh ? 8 + i : i);
      O[(size_t)row * (NQ * HD) + h * HD + n * 16 + l16] = (bf16)(accO[n][i] / lrow[i]);
    }
}

// ---------------------------------------------------------------------------
extern "C" void kernel_launch(void* const* d_in, const int* in_sizes, int n_in,
                              void* d_out, int out_size, void* d_ws, size_t ws_size,
                              hipStream_t stream) {
  const float* hs   = (const float*)d_in[1];
  const float* Wqkv = (const float*)d_in[2];
  const float* Wo   = (const float*)d_in[3];
  const float* qg   = (const float*)d_in[4];
  const float* kg   = (const float*)d_in[5];
  const float* cosb = (const float*)d_in[6];
  const float* sinb = (const float*)d_in[7];
  float* out = (float*)d_out;

  char* ws = (char*)d_ws;
  bf16*  hsb   = (bf16*)ws;  ws += (size_t)S_LEN * HID  * 2;
  bf16*  wqkvT = (bf16*)ws;  ws += (size_t)QKVD  * HID  * 2;
  bf16*  woT   = (bf16*)ws;  ws += (size_t)HID   * HID  * 2;
  float* qkv   = (float*)ws; ws += (size_t)S_LEN * QKVD * 4;
  bf16*  Qb    = (bf16*)ws;  ws += (size_t)S_LEN * NQ  * HD * 2;
  bf16*  Kb    = (bf16*)ws;  ws += (size_t)S_LEN * NKV * HD * 2;
  bf16*  Vtb   = (bf16*)ws;  ws += (size_t)NKV * HD * S_LEN * 2;
  bf16*  Ob    = (bf16*)ws;

  cvt_bf16<<<((size_t)S_LEN * HID) / 256, 256, 0, stream>>>(hs, hsb, (size_t)S_LEN * HID);
  transpose_cvt<<<dim3(QKVD / 32, HID / 32), dim3(32, 8), 0, stream>>>(Wqkv, wqkvT, HID, QKVD);
  transpose_cvt<<<dim3(HID  / 32, HID / 32), dim3(32, 8), 0, stream>>>(Wo,   woT,   HID, HID);

  gemm_bf16<<<dim3(QKVD / 128, S_LEN / 128), 256, 0, stream>>>(hsb, wqkvT, qkv, S_LEN, QKVD, HID);
  rms_rope<<<dim3(S_LEN, NQ + 2 * NKV), 128, 0, stream>>>(qkv, qg, kg, cosb, sinb, Qb, Kb, Vtb);
  flash_attn<<<dim3(S_LEN / 64, NQ), 128, 0, stream>>>(Qb, Kb, Vtb, Ob);
  gemm_bf16<<<dim3(HID / 128, S_LEN / 128), 256, 0, stream>>>(Ob, woT, out, S_LEN, HID, HID);
}
